// AlgebraicTransformerLM_77695958384933
// MI455X (gfx1250) — matrix operand order repaired and checked by reading, not compiled, not run on hardware
//
#include <hip/hip_runtime.h>

// ---------------------------------------------------------------------------
// AlgebraicTransformerLM forward for gfx1250 (MI455X).
// bf16 WMMA GEMMs, TDM (tensor_load_to_lds) double-buffered tile staging,
// fused streaming attention with WMMA-computed row denominators.
// B=2 T=1024 V=32000 D=1024 H=16 DH=64 L=4 DFF=2730 (K-padded to 2752).
// Workspace requirement: ~256 MB.
// ---------------------------------------------------------------------------

namespace {

constexpr int kB   = 2;
constexpr int kT   = 1024;
constexpr int kV   = 32000;
constexpr int kD   = 1024;
constexpr int kH   = 16;
constexpr int kL   = 4;
constexpr int kDH  = 64;
constexpr int kDFF = 2730;          // int(1024*8/3)
constexpr int kDFFP = 2752;         // padded to multiple of 64 (86*32)
constexpr int kMR  = kB * kT;       // 2048 token rows
constexpr float kSCALE = 0.125f;    // 1/sqrt(64)
constexpr float kEPS = 1e-6f;

typedef __attribute__((ext_vector_type(16))) __bf16 bfrag_t;  // A/B operand
typedef __attribute__((ext_vector_type(8)))  float  ffrag_t;  // C/D operand

union FragU {
    bfrag_t v;
    uint4   q[2];
    unsigned short u[16];
};

__device__ inline unsigned short cvt_bf16(float f) {
    unsigned int x = __builtin_bit_cast(unsigned int, f);
    unsigned int r = x + 0x7FFFu + ((x >> 16) & 1u);   // round-to-nearest-even
    return (unsigned short)(r >> 16);
}

__device__ inline ffrag_t wmma_bf16(const FragU& a, const FragU& b, ffrag_t c) {
    // D = A(16x32 bf16) * B(32x16 bf16) + C(16x16 f32)
    return __builtin_amdgcn_wmma_f32_16x16x32_bf16(
        false, a.v, false, b.v, (short)0, c, false, false);
}

__device__ inline float rational_sigmoid(float x) {
    return 0.5f * (x / (fabsf(x) + 1.0f) + 1.0f);
}

} // namespace

// ---------------------------------------------------------------------------
// TDM: Tensor Data Mover 2-D tile load (global -> LDS), CDNA5 D# descriptor.
// This toolchain exposes the 6-arg builtin:
//   (uint32x4 g0, int32x8 g1, int32x4 g2, int32x4 g3, int32x8 g4, i32 cpol)
// ---------------------------------------------------------------------------
#if __has_builtin(__builtin_amdgcn_tensor_load_to_lds) && \
    __has_builtin(__builtin_amdgcn_s_wait_tensorcnt)
#define GEMM_USE_TDM 1
#else
#define GEMM_USE_TDM 0
#endif

#if GEMM_USE_TDM
typedef unsigned int tdm4u __attribute__((ext_vector_type(4)));
typedef int          tdm8i __attribute__((ext_vector_type(8)));
typedef int          tdm4i __attribute__((ext_vector_type(4)));

// Load a (tile_d1 rows x tile_d0 elems) tile of 2-byte elements into LDS.
// tensor dims are "remaining extent" from gaddr; OOB rows read as zero.
__device__ inline void tdm_load_2d(unsigned lds_off, const void* gaddr,
                                   unsigned tile_d0, unsigned tile_d1,
                                   unsigned td0, unsigned td1,
                                   unsigned stride0 /* elems */) {
    const unsigned long long ga = (unsigned long long)gaddr;
    tdm4u g0;
    g0[0] = 1u;                                   // count=1, user mode
    g0[1] = lds_off;                              // LDS byte address
    g0[2] = (unsigned)ga;                         // global addr [31:0]
    g0[3] = (unsigned)(ga >> 32) | 0x80000000u;   // [56:32] | type=2
    tdm8i g1;
    g1[0] = 0x00010000;                           // data_size=1 (2 bytes)
    g1[1] = (int)((td0 & 0xFFFFu) << 16);         // tensor_dim0[15:0] @bit48
    g1[2] = (int)((td0 >> 16) | ((td1 & 0xFFFFu) << 16));
    g1[3] = (int)((td1 >> 16) | (tile_d0 << 16)); // tile_dim0 @bit112
    g1[4] = (int)tile_d1;                         // tile_dim1 (tile_dim2=0)
    g1[5] = (int)stride0;                         // tensor_dim0_stride[31:0]
    g1[6] = 0;                                    // stride0[47:32]=0, stride1=0
    g1[7] = 0;
    const tdm4i z4 = {0, 0, 0, 0};
    const tdm8i z8 = {0, 0, 0, 0, 0, 0, 0, 0};
    __builtin_amdgcn_tensor_load_to_lds(g0, g1, z4, z4, z8, 0);
}
#endif

// ---------------------------------------------------------------------------
// Elementwise f32 -> bf16 conversion (grid-stride)
// ---------------------------------------------------------------------------
__global__ void cvt_kernel(const float* __restrict__ in,
                           unsigned short* __restrict__ out, size_t n) {
    for (size_t i = (size_t)blockIdx.x * blockDim.x + threadIdx.x;
         i < n; i += (size_t)gridDim.x * blockDim.x)
        out[i] = cvt_bf16(in[i]);
}

// w3_w (L, D, DFF) f32 -> (L, D, DFFP) bf16, zero padding in K tail
__global__ void w3pad_kernel(const float* __restrict__ w3,
                             unsigned short* __restrict__ out) {
    const size_t total = (size_t)kL * kD * kDFFP;
    for (size_t i = (size_t)blockIdx.x * blockDim.x + threadIdx.x;
         i < total; i += (size_t)gridDim.x * blockDim.x) {
        size_t row = i / kDFFP;
        int c = (int)(i % kDFFP);
        out[i] = (c < kDFF) ? cvt_bf16(w3[row * (size_t)kDFF + c]) : (unsigned short)0;
    }
}

// x = emb[input_ids]  (f32 residual stream)
__global__ void embed_kernel(const int* __restrict__ ids,
                             const float* __restrict__ emb,
                             float* __restrict__ x) {
    const size_t total = (size_t)kMR * kD;
    for (size_t i = (size_t)blockIdx.x * blockDim.x + threadIdx.x;
         i < total; i += (size_t)gridDim.x * blockDim.x) {
        size_t row = i / kD;
        int c = (int)(i % kD);
        x[i] = emb[(size_t)ids[row] * kD + c];
    }
}

// mean_error_norm: y_bf16 = x / (mean|x| + eps) * w      (one block per row)
__global__ __launch_bounds__(256) void norm_kernel(const float* __restrict__ x,
                                                   const float* __restrict__ w,
                                                   unsigned short* __restrict__ y) {
    const int row = blockIdx.x;
    const float* xr = x + (size_t)row * kD;
    float s = 0.0f;
    for (int i = threadIdx.x; i < kD; i += 256) s += fabsf(xr[i]);
    for (int off = 16; off > 0; off >>= 1) s += __shfl_xor(s, off, 32);
    __shared__ float red[8];
    const int lane = threadIdx.x & 31, wv = threadIdx.x >> 5;
    if (lane == 0) red[wv] = s;
    __syncthreads();
    float tot = 0.0f;
#pragma unroll
    for (int i = 0; i < 8; ++i) tot += red[i];
    const float inv = 1.0f / (tot / (float)kD + kEPS);
    unsigned short* yr = y + (size_t)row * kD;
    for (int i = threadIdx.x; i < kD; i += 256)
        yr[i] = cvt_bf16(xr[i] * inv * w[i]);
}

// swiglu activation: hid = gate * rs(gate) * val  (bf16, zero K-padding)
__global__ void act_kernel(const float* __restrict__ gate,
                           const float* __restrict__ val,
                           unsigned short* __restrict__ hid) {
    const size_t total = (size_t)kMR * kDFFP;
    for (size_t i = (size_t)blockIdx.x * blockDim.x + threadIdx.x;
         i < total; i += (size_t)gridDim.x * blockDim.x) {
        int c = (int)(i % kDFFP);
        if (c < kDFF) {
            float g = gate[i];
            hid[i] = cvt_bf16(g * rational_sigmoid(g) * val[i]);
        } else {
            hid[i] = 0;
        }
    }
}

// V^T materialization: vt[(b*H+h)*64 + d][s] = v[b,s,h,d]  (tiled via LDS)
__global__ __launch_bounds__(256) void vtrans_kernel(const unsigned short* __restrict__ qkv,
                                                     unsigned short* __restrict__ vt) {
    __shared__ unsigned short tile[32][33];
    const int sb = blockIdx.x * 32;           // key/pos tile
    const int g  = blockIdx.y;                // (b*H+h)*2 + d-half
    const int d0 = (g & 1) * 32;
    const int bh = g >> 1;
    const int b = bh >> 4, h = bh & 15;
    const int c  = threadIdx.x & 31;
    const int r0 = threadIdx.x >> 5;          // 8 rows per pass
#pragma unroll
    for (int rr = 0; rr < 32; rr += 8)
        tile[rr + r0][c] =
            qkv[(size_t)(b * kT + sb + rr + r0) * (3 * kD) + 2 * kD + h * kDH + d0 + c];
    __syncthreads();
    unsigned short* vtb = vt + ((size_t)bh * kDH + d0) * kT;
#pragma unroll
    for (int rr = 0; rr < 32; rr += 8)
        vtb[(size_t)(rr + r0) * kT + sb + c] = tile[c][rr + r0];
}

// ---------------------------------------------------------------------------
// Tiled WMMA GEMM:  C(MxN) = A(MxK bf16, lda==K) * W(NxK bf16)^T
// 128 threads = 4 waves; block tile 64(M)x128(N); each wave owns 32x64
// (= 8 WMMA accumulators). Tiles staged into double-buffered LDS by the
// Tensor Data Mover (wave 0 issues 2-D D# descriptors, TENSORcnt waits),
// falling back to synchronous loads if the TDM builtin is unavailable.
// EPI: 0 = f32 store, 1 = f32 "+=", 2 = bf16 store.
// ---------------------------------------------------------------------------
template <int EPI>
__global__ __launch_bounds__(128) void gemm_kernel(const unsigned short* __restrict__ A,
                                                   const unsigned short* __restrict__ W,
                                                   void* __restrict__ Cp,
                                                   int M, int N, int K,
                                                   int lda, int ldc) {
    __shared__ unsigned short sA[2][64 * 32];   // 2 x 4 KB
    __shared__ unsigned short sB[2][128 * 32];  // 2 x 8 KB
    const int tid  = threadIdx.x;
    const int lane = tid & 31;
    const int waveu = __builtin_amdgcn_readfirstlane(tid >> 5);  // scalar wave id
    const int wm = waveu >> 1, wn = waveu & 1;
    const int col = lane & 15, hi = lane >> 4;
    const int m0 = blockIdx.y * 64, n0 = blockIdx.x * 128;
    const int nsteps = K / 32;

    ffrag_t acc[2][4];
#pragma unroll
    for (int i = 0; i < 2; ++i)
#pragma unroll
        for (int j = 0; j < 4; ++j) acc[i][j] = (ffrag_t){0, 0, 0, 0, 0, 0, 0, 0};

#if GEMM_USE_TDM
    const unsigned sA_off = (unsigned)(size_t)&sA[0][0];
    const unsigned sB_off = (unsigned)(size_t)&sB[0][0];
    auto issue = [&](int buf, int k0) {
        tdm_load_2d(sA_off + (unsigned)buf * (64 * 32 * 2),
                    A + (size_t)m0 * lda + k0,
                    32, 64, (unsigned)(K - k0), (unsigned)(M - m0), (unsigned)lda);
        tdm_load_2d(sB_off + (unsigned)buf * (128 * 32 * 2),
                    W + (size_t)n0 * K + k0,
                    32, 128, (unsigned)(K - k0), (unsigned)(N - n0), (unsigned)K);
    };
    if (waveu == 0) issue(0, 0);
#endif

    for (int s = 0; s < nsteps; ++s) {
        const int buf = s & 1;
        const int k0 = s * 32;
#if GEMM_USE_TDM
        if (waveu == 0) {
            if (s + 1 < nsteps) {
                issue(buf ^ 1, k0 + 32);                  // prefetch next stage
                __builtin_amdgcn_s_wait_tensorcnt(2);     // current stage done
            } else {
                __builtin_amdgcn_s_wait_tensorcnt(0);
            }
        }
        __syncthreads();
#else
        {
            const unsigned short* ap =
                A + (size_t)(m0 + (tid >> 1)) * lda + k0 + (tid & 1) * 16;
            *(uint4*)&sA[buf][(tid >> 1) * 32 + (tid & 1) * 16] = *(const uint4*)ap;
            int nr = n0 + tid; if (nr > N - 1) nr = N - 1;
            const unsigned short* bp = W + (size_t)nr * K + k0;
#pragma unroll
            for (int c = 0; c < 4; ++c)
                *(uint4*)&sB[buf][tid * 32 + c * 8] = *(const uint4*)(bp + c * 8);
        }
        __syncthreads();
#endif
        FragU a[2], b[4];
#pragma unroll
        for (int i = 0; i < 2; ++i) {
            // A frag: lane=row, K chunks {hi*8..+7, hi*8+16..+23}
            const unsigned short* p = &sA[buf][(wm * 32 + i * 16 + col) * 32 + hi * 8];
            a[i].q[0] = *(const uint4*)p;
            a[i].q[1] = *(const uint4*)(p + 16);
        }
#pragma unroll
        for (int j = 0; j < 4; ++j) {
            // B frag: lane=col of B (= row of W), K chunks {hi*8..+7, hi*8+16..+23}
            const unsigned short* p = &sB[buf][(wn * 64 + j * 16 + col) * 32 + hi * 8];
            b[j].q[0] = *(const uint4*)p;
            b[j].q[1] = *(const uint4*)(p + 16);
        }
#pragma unroll
        for (int i = 0; i < 2; ++i)
#pragma unroll
            for (int j = 0; j < 4; ++j)
                acc[i][j] = wmma_bf16(a[i], b[j], acc[i][j]);
        __syncthreads();   // protect buffer reuse before next DMA/stores
    }

#pragma unroll
    for (int i = 0; i < 2; ++i)
#pragma unroll
        for (int j = 0; j < 4; ++j)
#pragma unroll
            for (int r = 0; r < 8; ++r) {
                const int m = m0 + wm * 32 + i * 16 + r + hi * 8;
                const int n = n0 + wn * 64 + j * 16 + col;
                if (n < N) {
                    if (EPI == 0)
                        ((float*)Cp)[(size_t)m * ldc + n] = acc[i][j][r];
                    else if (EPI == 1)
                        ((float*)Cp)[(size_t)m * ldc + n] += acc[i][j][r];
                    else
                        ((unsigned short*)Cp)[(size_t)m * ldc + n] = cvt_bf16(acc[i][j][r]);
                }
            }
}

// ---------------------------------------------------------------------------
// Fused attention. grid = (T/64, B*H); 4 waves, each owns 16 query rows.
// Streaming over 32-key tiles: scores = q@k^T (WMMA), ALiBi + causal mask +
// rational_sigmoid^4 (branchless), accumulate s4@V (WMMA, V^T pre-staged in
// global) and the row denominator via WMMA against an all-ones B fragment.
// All wave-uniform control flow is scalarized (readfirstlane) so EXEC stays
// all-ones across every WMMA.
// ---------------------------------------------------------------------------
__global__ __launch_bounds__(128) void attn_kernel(const unsigned short* __restrict__ qkv,
                                                   const unsigned short* __restrict__ vt,
                                                   unsigned short* __restrict__ ctx) {
    __shared__ unsigned short pshm[4][16 * 32];   // per-wave unnormalized probs

    const int tid  = threadIdx.x;
    const int lane = tid & 31;
    const int wave = __builtin_amdgcn_readfirstlane(tid >> 5);   // scalar
    const int col = lane & 15, hi = lane >> 4;
    const int bh = blockIdx.y, b = bh >> 4, h = bh & 15;
    const int t_base = blockIdx.x * 64 + wave * 16;              // scalar
    const int t_end = t_base + 15;                               // scalar
    const float slope = exp2f(-0.5f * (float)(h + 1));
    const int rowstride = 3 * kD;

    // q A-fragments for K (=d) steps 0..31 and 32..63
    FragU a0, a1;
    {
        const unsigned short* qp =
            qkv + (size_t)(b * kT + t_base + col) * rowstride + h * kDH;
        a0.q[0] = *(const uint4*)(qp + hi * 8);
        a0.q[1] = *(const uint4*)(qp + hi * 8 + 16);
        a1.q[0] = *(const uint4*)(qp + 32 + hi * 8);
        a1.q[1] = *(const uint4*)(qp + 32 + hi * 8 + 16);
    }
    FragU ones;
#pragma unroll
    for (int i = 0; i < 16; ++i) ones.u[i] = 0x3F80;   // bf16 1.0

    ffrag_t o[4];
#pragma unroll
    for (int j = 0; j < 4; ++j) o[j] = (ffrag_t){0, 0, 0, 0, 0, 0, 0, 0};
    ffrag_t dacc = (ffrag_t){0, 0, 0, 0, 0, 0, 0, 0};

    unsigned short* pw = pshm[wave];
    const unsigned short* vtb = vt + (size_t)bh * kDH * kT;

    for (int s0 = 0; s0 <= t_end; s0 += 32) {
        // --- scores for two 16-key subtiles -> s4 -> pshm (A-frag layout) ---
#pragma unroll
        for (int sub = 0; sub < 2; ++sub) {
            const int sb = s0 + sub * 16;        // scalar
            if (sb <= t_end) {                   // scalar branch
                const int skey = sb + col;       // this lane's key (column)
                const unsigned short* kp =
                    qkv + (size_t)(b * kT + skey) * rowstride + kD + h * kDH;
                FragU kb0, kb1;
                kb0.q[0] = *(const uint4*)(kp + hi * 8);
                kb0.q[1] = *(const uint4*)(kp + hi * 8 + 16);
                kb1.q[0] = *(const uint4*)(kp + 32 + hi * 8);
                kb1.q[1] = *(const uint4*)(kp + 32 + hi * 8 + 16);
                ffrag_t sc = (ffrag_t){0, 0, 0, 0, 0, 0, 0, 0};
                sc = wmma_bf16(a0, kb0, sc);
                sc = wmma_bf16(a1, kb1, sc);
#pragma unroll
                for (int r = 0; r < 8; ++r) {
                    const int tq = t_base + r + hi * 8;
                    const float xs = sc[r] * kSCALE - slope * (float)(tq - skey);
                    const float rs = rational_sigmoid(xs);
                    const float rs2 = rs * rs;
                    const float p4 = (skey <= tq) ? rs2 * rs2 : 0.0f;  // branchless
                    pw[(r + hi * 8) * 32 + sub * 16 + col] = cvt_bf16(p4);
                }
            } else {
#pragma unroll
                for (int r = 0; r < 8; ++r)
                    pw[(r + hi * 8) * 32 + sub * 16 + col] = 0;
            }
        }
        // wave-private LDS, cross-lane: ensure stores landed before frag read
        asm volatile("s_wait_dscnt 0x0" ::: "memory");

        // --- P as A-frag (16x32) ---
        FragU pf;
        {
            const unsigned short* pp = &pw[col * 32 + hi * 8];
            pf.q[0] = *(const uint4*)pp;
            pf.q[1] = *(const uint4*)(pp + 16);
        }
        // --- o += P @ V (from global V^T),  dacc += P @ ones ---
#pragma unroll
        for (int j = 0; j < 4; ++j) {
            FragU vb;
            const unsigned short* vpp = vtb + (size_t)(j * 16 + col) * kT + s0 + hi * 8;
            vb.q[0] = *(const uint4*)vpp;
            vb.q[1] = *(const uint4*)(vpp + 16);
            o[j] = wmma_bf16(pf, vb, o[j]);
        }
        dacc = wmma_bf16(pf, ones, dacc);
    }

    // --- normalize + store context (bf16, (B*T, D) with head offset) ---
#pragma unroll
    for (int j = 0; j < 4; ++j)
#pragma unroll
        for (int r = 0; r < 8; ++r) {
            const int tq = t_base + r + hi * 8;
            const float dv = o[j][r] / (dacc[r] + kEPS);
            ctx[(size_t)(b * kT + tq) * kD + h * kDH + j * 16 + col] = cvt_bf16(dv);
        }
}

// ---------------------------------------------------------------------------
// Host orchestration
// ---------------------------------------------------------------------------
extern "C" void kernel_launch(void* const* d_in, const int* in_sizes, int n_in,
                              void* d_out, int out_size, void* d_ws, size_t ws_size,
                              hipStream_t stream) {
    (void)in_sizes; (void)n_in; (void)out_size; (void)ws_size;

    const int*   ids   = (const int*)  d_in[0];
    const float* emb   = (const float*)d_in[1];
    const float* qkv_w = (const float*)d_in[2];
    const float* out_w = (const float*)d_in[3];
    const float* n1_w  = (const float*)d_in[4];
    const float* n2_w  = (const float*)d_in[5];
    const float* wm_w  = (const float*)d_in[6];
    const float* w3_w  = (const float*)d_in[7];
    const float* fn_w  = (const float*)d_in[8];
    float* logits = (float*)d_out;

    // workspace carving (256B aligned). Total ~256 MB.
    char* ws = (char*)d_ws;
    size_t off = 0;
    auto carve = [&](size_t bytes) -> void* {
        void* p = ws + off;
        off += (bytes + 255) & ~(size_t)255;
        return p;
    };
    float*          xf     = (float*)         carve((size_t)kMR * kD * 4);
    unsigned short* ybf    = (unsigned short*)carve((size_t)kMR * kD * 2);
    unsigned short* qkvbf  = (unsigned short*)carve((size_t)kMR * 3 * kD * 2);
    unsigned short* ctxbf  = (unsigned short*)carve((size_t)kMR * kD * 2);
    unsigned short* vtbf   = (unsigned short*)carve((size_t)kB * kH * kDH * kT * 2);
    float*          gatef  = (float*)         carve((size_t)kMR * kDFFP * 4);
    float*          valf   = (float*)         carve((size_t)kMR * kDFFP * 4);
    unsigned short* hidbf  = (unsigned short*)carve((size_t)kMR * kDFFP * 2);
    unsigned short* embbf  = (unsigned short*)carve((size_t)kV * kD * 2);
    unsigned short* qkvwbf = (unsigned short*)carve((size_t)kL * 3 * kD * kD * 2);
    unsigned short* outwbf = (unsigned short*)carve((size_t)kL * kD * kD * 2);
    unsigned short* wmwbf  = (unsigned short*)carve((size_t)kL * 2 * kDFF * kD * 2);
    unsigned short* w3pbf  = (unsigned short*)carve((size_t)kL * kD * kDFFP * 2);

    // --- weight conversions (every call; deterministic, no cached state) ---
    cvt_kernel<<<4096, 256, 0, stream>>>(emb,   embbf,  (size_t)kV * kD);
    cvt_kernel<<<2048, 256, 0, stream>>>(qkv_w, qkvwbf, (size_t)kL * 3 * kD * kD);
    cvt_kernel<<<1024, 256, 0, stream>>>(out_w, outwbf, (size_t)kL * kD * kD);
    cvt_kernel<<<2048, 256, 0, stream>>>(wm_w,  wmwbf,  (size_t)kL * 2 * kDFF * kD);
    w3pad_kernel<<<2048, 256, 0, stream>>>(w3_w, w3pbf);

    // --- embedding ---
    embed_kernel<<<2048, 256, 0, stream>>>(ids, emb, xf);

    const dim3 blk(128);
    for (int l = 0; l < kL; ++l) {
        // attention
        norm_kernel<<<kMR, 256, 0, stream>>>(xf, n1_w + (size_t)l * kD, ybf);
        gemm_kernel<2><<<dim3(24, 32), blk, 0, stream>>>(
            ybf, qkvwbf + (size_t)l * 3 * kD * kD, qkvbf,
            kMR, 3 * kD, kD, kD, 3 * kD);
        vtrans_kernel<<<dim3(kT / 32, kB * kH * 2), 256, 0, stream>>>(qkvbf, vtbf);
        attn_kernel<<<dim3(kT / 64, kB * kH), blk, 0, stream>>>(qkvbf, vtbf, ctxbf);
        gemm_kernel<1><<<dim3(8, 32), blk, 0, stream>>>(
            ctxbf, outwbf + (size_t)l * kD * kD, xf,
            kMR, kD, kD, kD, kD);

        // swiglu MLP
        norm_kernel<<<kMR, 256, 0, stream>>>(xf, n2_w + (size_t)l * kD, ybf);
        const unsigned short* wml = wmwbf + (size_t)l * 2 * kDFF * kD;
        gemm_kernel<0><<<dim3(22, 32), blk, 0, stream>>>(
            ybf, wml, gatef, kMR, kDFF, kD, kD, kDFFP);
        gemm_kernel<0><<<dim3(22, 32), blk, 0, stream>>>(
            ybf, wml + (size_t)kDFF * kD, valf, kMR, kDFF, kD, kD, kDFFP);
        act_kernel<<<4096, 256, 0, stream>>>(gatef, valf, hidbf);
        gemm_kernel<1><<<dim3(8, 32), blk, 0, stream>>>(
            hidbf, w3pbf + (size_t)l * kD * kDFFP, xf,
            kMR, kD, kDFFP, kDFFP, kD);
    }

    // final norm + logits = y @ emb^T
    norm_kernel<<<kMR, 256, 0, stream>>>(xf, fn_w, ybf);
    gemm_kernel<0><<<dim3(250, 32), blk, 0, stream>>>(
        ybf, embbf, logits, kMR, kV, kD, kD, kV);
}
